// GradMimicryNetwork_72000831750294
// MI455X (gfx1250) — compile-verified, matrix operand order checked by reading, not run on hardware
//
#include <hip/hip_runtime.h>
#include <hip/hip_bf16.h>
#include <stdint.h>

typedef _Float16 v16h __attribute__((ext_vector_type(16)));
typedef _Float16 v8h  __attribute__((ext_vector_type(8)));
typedef float    v8f  __attribute__((ext_vector_type(8)));

// ---------------- problem constants ----------------
#define T_STEPS 200
#define NBATCH  256
// layer sizes (HO padded 10->16), K0 padded 784->800
#define H0 256
#define H1 128
#define H2 64
#define H3 32
#define HOP 16
#define K0P 800

// ---------------- workspace layout (bytes, all 16B aligned) ----------------
// I0   : (T*B=51200) x 256 f32   precomputed layer-0 input currents
// rwXt : relu(wX) as f16, transposed [out][in], K padded to mult of 32
// V    : membrane potentials f32 (v0..v3 + padded output v4)
// S    : double-buffered spike state f16 (s0..s3)
// At   : transposed input spikes f16, rows r = t*256+b, K padded to 800
static constexpr size_t OFF_I0  = 0;
static constexpr size_t SZ_I0   = (size_t)51200 * 256 * 4;          // 52,428,800
static constexpr size_t OFF_RW0 = OFF_I0 + SZ_I0;
static constexpr size_t SZ_RW0  = (size_t)H0 * K0P * 2;             // 409,600
static constexpr size_t OFF_RW1 = OFF_RW0 + SZ_RW0;
static constexpr size_t SZ_RW1  = (size_t)H1 * H0 * 2;              // 65,536
static constexpr size_t OFF_RW2 = OFF_RW1 + SZ_RW1;
static constexpr size_t SZ_RW2  = (size_t)H2 * H1 * 2;              // 16,384
static constexpr size_t OFF_RW3 = OFF_RW2 + SZ_RW2;
static constexpr size_t SZ_RW3  = (size_t)H3 * H2 * 2;              // 4,096
static constexpr size_t OFF_RW4 = OFF_RW3 + SZ_RW3;
static constexpr size_t SZ_RW4  = (size_t)HOP * H3 * 2;             // 1,024
static constexpr size_t OFF_V   = OFF_RW4 + SZ_RW4;
static constexpr size_t NV      = (size_t)NBATCH * (H0 + H1 + H2 + H3 + HOP); // 126,976
static constexpr size_t SZ_V    = NV * 4;                           // 507,904
static constexpr size_t OFF_S   = OFF_V + SZ_V;
static constexpr size_t SBUF    = (size_t)NBATCH * (H0 + H1 + H2 + H3);       // 122,880 halves
static constexpr size_t SZ_S    = 2 * SBUF * 2;                     // 491,520
static constexpr size_t OFF_AT  = OFF_S + SZ_S;
static constexpr size_t SZ_AT   = (size_t)51200 * K0P * 2;          // 81,920,000

// output offsets (f32 elements, concat order of reference return tuple)
static constexpr size_t O_H0 = 0;
static constexpr size_t O_H1 = O_H0 + (size_t)NBATCH * H0 * T_STEPS;  // 13,107,200
static constexpr size_t O_H2 = O_H1 + (size_t)NBATCH * H1 * T_STEPS;  // 19,660,800
static constexpr size_t O_H3 = O_H2 + (size_t)NBATCH * H2 * T_STEPS;  // 22,937,600
static constexpr size_t O_OS = O_H3 + (size_t)NBATCH * H3 * T_STEPS;  // 24,576,000
static constexpr size_t O_FR = O_OS + (size_t)NBATCH * 10 * T_STEPS;  // 25,088,000

// ---------------- WMMA fragment loader ----------------
// 16-bit A 16x32 layout (ISA 7.12.2): lane L -> row M=L&15; per lane two
// contiguous 8-half runs at K = (L>>4)*8 and K = 16+(L>>4)*8.
// B is stored pre-transposed [N][K] so the same loader serves both operands.
__device__ __forceinline__ v16h load_frag16(const _Float16* __restrict__ base,
                                            int ld, int lane) {
  const int r  = lane & 15;
  const int hf = lane >> 4;
  const _Float16* p = base + (size_t)r * ld + hf * 8;
  v8h lo = *(const v8h*)(p);       // K = hf*8 .. hf*8+7
  v8h hi = *(const v8h*)(p + 16);  // K = 16+hf*8 .. 16+hf*8+7
  v16h f;
#pragma unroll
  for (int i = 0; i < 8; ++i) { f[i] = lo[i]; f[i + 8] = hi[i]; }
  return f;
}

// ---------------- fused 16x16 GEMM tile + LIF update ----------------
// current = A(spikes_prev, BxK f16) @ Bw([N][K] f16); then LIF on the tile.
// v_new = 0.5*v + I ; s = (v_new >= 1) ; v = s ? 0 : v_new
__device__ __forceinline__ void gemm_lif_tile(
    const _Float16* __restrict__ A, const _Float16* __restrict__ Bw, int K,
    int m0, int n0, int ldH, float* __restrict__ vstate,
    _Float16* __restrict__ snext, float* __restrict__ outp, int Hvalid,
    int t, int lane) {
  v8f acc = {};
  const _Float16* a_base = A + (size_t)m0 * K;
  const _Float16* b_base = Bw + (size_t)n0 * K;
  for (int kk = 0; kk < K; kk += 32) {
    v16h a = load_frag16(a_base + kk, K, lane);
    v16h b = load_frag16(b_base + kk, K, lane);
    acc = __builtin_amdgcn_wmma_f32_16x16x32_f16(false, a, false, b,
                                                 (short)0, acc, false, false);
  }
  const int n  = n0 + (lane & 15);
  const int hf = lane >> 4;
#pragma unroll
  for (int i = 0; i < 8; ++i) {
    const int m   = m0 + hf * 8 + i;
    const int idx = m * ldH + n;
    float v  = vstate[idx];
    float vn = 0.5f * v + acc[i];
    float s  = (vn - 1.0f >= 0.0f) ? 1.0f : 0.0f;
    vstate[idx] = (s > 0.0f) ? 0.0f : vn;
    if (snext) snext[idx] = (_Float16)s;
    if (n < Hvalid)
      outp[(size_t)m * Hvalid * T_STEPS + (size_t)n * T_STEPS + t] = s;
  }
}

// ---------------- layer-0 LIF (currents precomputed in I0) ----------------
__device__ __forceinline__ void lif0(const float* __restrict__ I0t,
                                     float* __restrict__ v0,
                                     _Float16* __restrict__ s0n,
                                     float* __restrict__ out0, int unit,
                                     int t, int lane) {
  const int base = unit * 256 + lane;
#pragma unroll
  for (int j = 0; j < 8; ++j) {
    const int e = base + j * 32;     // e = b*256 + n, coalesced
    float vn = 0.5f * v0[e] + I0t[e];
    float s  = (vn - 1.0f >= 0.0f) ? 1.0f : 0.0f;
    v0[e]  = (s > 0.0f) ? 0.0f : vn;
    s0n[e] = (_Float16)s;
    const int b = e >> 8, n = e & 255;
    out0[(size_t)b * H0 * T_STEPS + (size_t)n * T_STEPS + t] = s;
  }
}

// ================= kernels =================

// zero membrane potentials + both spike-state buffers
__global__ void init_state_kernel(char* ws) {
  size_t id = (size_t)blockIdx.x * blockDim.x + threadIdx.x;
  size_t nwords = (SZ_V + SZ_S) / 4;
  if (id < nwords) ((uint32_t*)(ws + OFF_V))[id] = 0u;
}

// relu(w) -> f16, transposed [N][K], K padded with zeros
__global__ void prep_weights_kernel(const float* __restrict__ w,
                                    _Float16* __restrict__ dst, int Kr, int Kp,
                                    int Nr, int Np) {
  int id = blockIdx.x * blockDim.x + threadIdx.x;
  if (id >= Kp * Np) return;
  int k = id % Kp, n = id / Kp;
  float v = 0.0f;
  if (k < Kr && n < Nr) { v = w[(size_t)k * Nr + n]; v = v > 0.0f ? v : 0.0f; }
  dst[(size_t)n * Kp + k] = (_Float16)v;
}

// input_spikes (B,784,T) f32 -> At (T*B, 800) f16, rows r = t*256 + b.
// Input (160 MB) is L2-resident on MI455X (192 MB L2), writes are coalesced.
__global__ void transpose_spikes_kernel(const float* __restrict__ in,
                                        char* ws) {
  size_t id = (size_t)blockIdx.x * blockDim.x + threadIdx.x;
  if (id >= (size_t)51200 * K0P) return;
  int k = (int)(id % K0P);
  size_t r = id / K0P;
  int t = (int)(r / NBATCH), b = (int)(r % NBATCH);
  float v = (k < 784) ? in[(size_t)b * 784 * T_STEPS + (size_t)k * T_STEPS + t]
                      : 0.0f;
  ((_Float16*)(ws + OFF_AT))[id] = (_Float16)v;
}

// big hoisted GEMM: I0 = At(51200x800) @ rw0t -> f32. Each wave: 16x64 strip
// (1 A-frag load amortized over 4 B-frags + 4 WMMAs per k-step, 25 k-steps).
__global__ void __launch_bounds__(128) gemm0_kernel(char* ws) {
  const int lane = threadIdx.x & 31;
  const int wid  = blockIdx.x * 4 + (threadIdx.x >> 5);  // 0..12799
  const int tm   = wid >> 2;                              // 3200 M-tiles
  const int tn4  = wid & 3;                               // 4 N-groups
  const _Float16* At  = (const _Float16*)(ws + OFF_AT);
  const _Float16* rw0 = (const _Float16*)(ws + OFF_RW0);
  float* I0 = (float*)(ws + OFF_I0);
  const int m0 = tm * 16;
  v8f acc[4] = {};
  const _Float16* a_base = At + (size_t)m0 * K0P;
  for (int kk = 0; kk < K0P; kk += 32) {
    v16h a = load_frag16(a_base + kk, K0P, lane);
#pragma unroll
    for (int q = 0; q < 4; ++q) {
      const int n0 = (tn4 * 4 + q) * 16;
      v16h b = load_frag16(rw0 + (size_t)n0 * K0P + kk, K0P, lane);
      acc[q] = __builtin_amdgcn_wmma_f32_16x16x32_f16(false, a, false, b,
                                                      (short)0, acc[q],
                                                      false, false);
    }
  }
  const int nl = lane & 15, hf = lane >> 4;
#pragma unroll
  for (int q = 0; q < 4; ++q) {
    const int n0 = (tn4 * 4 + q) * 16;
#pragma unroll
    for (int i = 0; i < 8; ++i) {
      const int m = m0 + hf * 8 + i;
      I0[(size_t)m * H0 + n0 + nl] = acc[q][i];
    }
  }
}

// one sequential timestep: 496 wave-units
//   u<256        : layer-0 LIF (currents from I0)
//   256..383     : layer-1 GEMM(K=256)+LIF   (128 tiles)
//   384..447     : layer-2 GEMM(K=128)+LIF   ( 64 tiles)
//   448..479     : layer-3 GEMM(K=64)+LIF    ( 32 tiles)
//   480..495     : output  GEMM(K=32)+LIF    ( 16 tiles, N valid=10)
__global__ void __launch_bounds__(128) step_kernel(char* ws,
                                                   float* __restrict__ out,
                                                   int t) {
  const int lane = threadIdx.x & 31;
  const int u = blockIdx.x * 4 + (threadIdx.x >> 5);

  float* I0 = (float*)(ws + OFF_I0);
  const _Float16* rw1 = (const _Float16*)(ws + OFF_RW1);
  const _Float16* rw2 = (const _Float16*)(ws + OFF_RW2);
  const _Float16* rw3 = (const _Float16*)(ws + OFF_RW3);
  const _Float16* rw4 = (const _Float16*)(ws + OFF_RW4);

  float* V  = (float*)(ws + OFF_V);
  float* v0 = V;
  float* v1 = v0 + (size_t)NBATCH * H0;
  float* v2 = v1 + (size_t)NBATCH * H1;
  float* v3 = v2 + (size_t)NBATCH * H2;
  float* v4 = v3 + (size_t)NBATCH * H3;

  _Float16* S  = (_Float16*)(ws + OFF_S);
  const int cur = t & 1;             // read buffer (holds s[t-1])
  _Float16* sc = S + (size_t)cur * SBUF;
  _Float16* sn = S + (size_t)(cur ^ 1) * SBUF;
  _Float16* s0c = sc;
  _Float16* s1c = s0c + (size_t)NBATCH * H0;
  _Float16* s2c = s1c + (size_t)NBATCH * H1;
  _Float16* s3c = s2c + (size_t)NBATCH * H2;
  _Float16* s0n = sn;
  _Float16* s1n = s0n + (size_t)NBATCH * H0;
  _Float16* s2n = s1n + (size_t)NBATCH * H1;
  _Float16* s3n = s2n + (size_t)NBATCH * H2;

  float* out0 = out + O_H0;
  float* out1 = out + O_H1;
  float* out2 = out + O_H2;
  float* out3 = out + O_H3;
  float* outO = out + O_OS;

  if (u < 256) {
    lif0(I0 + (size_t)t * NBATCH * H0, v0, s0n, out0, u, t, lane);
  } else if (u < 384) {
    const int tile = u - 256, tm = tile >> 3, tn = tile & 7;
    gemm_lif_tile(s0c, rw1, H0, tm * 16, tn * 16, H1, v1, s1n, out1, H1, t, lane);
  } else if (u < 448) {
    const int tile = u - 384, tm = tile >> 2, tn = tile & 3;
    gemm_lif_tile(s1c, rw2, H1, tm * 16, tn * 16, H2, v2, s2n, out2, H2, t, lane);
  } else if (u < 480) {
    const int tile = u - 448, tm = tile >> 1, tn = tile & 1;
    gemm_lif_tile(s2c, rw3, H2, tm * 16, tn * 16, H3, v3, s3n, out3, H3, t, lane);
  } else {
    const int tm = u - 480;
    gemm_lif_tile(s3c, rw4, H3, tm * 16, 0, HOP, v4, (_Float16*)nullptr, outO,
                  10, t, lane);
  }
}

// firing_rates[b,n] = mean_t output_spikes[b,n,t]
__global__ void rates_kernel(const float* __restrict__ outO,
                             float* __restrict__ rates) {
  int id = blockIdx.x * blockDim.x + threadIdx.x;
  if (id >= NBATCH * 10) return;
  const float* p = outO + (size_t)id * T_STEPS;
  float s = 0.0f;
  for (int t = 0; t < T_STEPS; ++t) s += p[t];
  rates[id] = s * (1.0f / T_STEPS);
}

// ================= launch =================
extern "C" void kernel_launch(void* const* d_in, const int* in_sizes, int n_in,
                              void* d_out, int out_size, void* d_ws,
                              size_t ws_size, hipStream_t stream) {
  const float* inp = (const float*)d_in[0];
  const float* w0  = (const float*)d_in[1];
  const float* w1  = (const float*)d_in[2];
  const float* w2  = (const float*)d_in[3];
  const float* w3  = (const float*)d_in[4];
  const float* w4  = (const float*)d_in[5];
  float* out = (float*)d_out;
  char*  ws  = (char*)d_ws;

  // zero v/s state (runs every call -> deterministic)
  {
    int nwords = (int)((SZ_V + SZ_S) / 4);
    init_state_kernel<<<(nwords + 255) / 256, 256, 0, stream>>>(ws);
  }
  // relu + f16 + transpose weights (N-major, K padded)
  prep_weights_kernel<<<(K0P * H0 + 255) / 256, 256, 0, stream>>>(
      w0, (_Float16*)(ws + OFF_RW0), 784, K0P, 256, H0);
  prep_weights_kernel<<<(H0 * H1 + 255) / 256, 256, 0, stream>>>(
      w1, (_Float16*)(ws + OFF_RW1), H0, H0, H1, H1);
  prep_weights_kernel<<<(H1 * H2 + 255) / 256, 256, 0, stream>>>(
      w2, (_Float16*)(ws + OFF_RW2), H1, H1, H2, H2);
  prep_weights_kernel<<<(H2 * H3 + 255) / 256, 256, 0, stream>>>(
      w3, (_Float16*)(ws + OFF_RW3), H2, H2, H3, H3);
  prep_weights_kernel<<<(H3 * HOP + 255) / 256, 256, 0, stream>>>(
      w4, (_Float16*)(ws + OFF_RW4), H3, H3, 10, HOP);

  // hoisted big GEMM path: transpose/convert spikes, then WMMA GEMM for I0
  transpose_spikes_kernel<<<(int)(((size_t)51200 * K0P + 255) / 256), 256, 0,
                            stream>>>(inp, ws);
  gemm0_kernel<<<3200, 128, 0, stream>>>(ws);

  // sequential LIF recurrence: 200 fused GEMM+LIF step kernels
  for (int t = 0; t < T_STEPS; ++t)
    step_kernel<<<124, 128, 0, stream>>>(ws, out, t);

  rates_kernel<<<(NBATCH * 10 + 255) / 256, 256, 0, stream>>>(out + O_OS,
                                                              out + O_FR);
}
